// DFA_9474697855653
// MI455X (gfx1250) — matrix operand order repaired and checked by reading, not compiled, not run on hardware
//
#include <hip/hip_runtime.h>
#include <hip/hip_bf16.h>

// Problem constants (from reference): N=1024 states, S=32 symbols, T=8192 steps.
#define NDIM 1024
#define SSYM 32
#define TSEQ 8192
#define NWG  16     // persistent workgroups in the chain kernel
#define BLK  256    // 8 waves of 32 (wave32 on gfx1250)

typedef __attribute__((ext_vector_type(16))) __bf16 v16bf;
typedef __attribute__((ext_vector_type(8)))  __bf16 v8bf;
typedef __attribute__((ext_vector_type(8)))  float  v8f;

// ---------------------------------------------------------------------------
// Workspace layout (d_ws):
//   [0, 64MB)              : DB  = bf16 softmaxed matrices, [S][N][N] row-major
//   [64MB, 64MB+8KB)       : qbuf = two f32 state vectors (ping-pong)
//   [64MB+8KB, +8B)        : barrier {count, phase}
// ---------------------------------------------------------------------------
#define DB_BYTES  ((size_t)SSYM * NDIM * NDIM * 2)

// ---------------------------------------------------------------------------
// Init: zero state vectors, set q0 = e_0, reset barrier. Deterministic per call.
// ---------------------------------------------------------------------------
__global__ void __launch_bounds__(BLK) dfa_init(float* qbuf, unsigned* bar) {
  int tid = threadIdx.x;
  for (int i = tid; i < 2 * NDIM; i += BLK) qbuf[i] = 0.0f;
  if (tid == 0) {
    qbuf[0] = 1.0f;   // q0 = e_0
    bar[0] = 0u;      // arrival count
    bar[1] = 0u;      // phase
  }
}

// ---------------------------------------------------------------------------
// Column softmax (axis=1 of [S,N,N]) + f32 -> bf16 convert.
// grid = (32 j-tiles, 32 symbols), block = 256 (tx = j in tile, ty = i group).
// Coalesced along j; online max/sum; two passes over delta (L2/HBM cheap).
// ---------------------------------------------------------------------------
__global__ void __launch_bounds__(BLK) dfa_softmax(const float* __restrict__ delta,
                                                   __bf16* __restrict__ DB) {
  const int s  = blockIdx.y;
  const int j0 = blockIdx.x * 32;
  const int tx = threadIdx.x & 31;   // column within tile
  const int ty = threadIdx.x >> 5;   // 0..7 row group

  const float* base = delta + (size_t)s * NDIM * NDIM;
  __bf16*      out  = DB    + (size_t)s * NDIM * NDIM;

  // online softmax accumulation over rows i = ty, ty+8, ...
  float m = -3.0e38f;
  float ssum = 0.0f;
  for (int i = ty; i < NDIM; i += 8) {
    float x  = base[(size_t)i * NDIM + j0 + tx];
    float nm = fmaxf(m, x);
    ssum = ssum * __expf(m - nm) + __expf(x - nm);
    m = nm;
  }

  __shared__ float lm[8][32];
  __shared__ float ls[8][32];
  lm[ty][tx] = m;
  ls[ty][tx] = ssum;
  __syncthreads();
  if (ty == 0) {
    float M = lm[0][tx], Ssum = ls[0][tx];
    #pragma unroll
    for (int r = 1; r < 8; ++r) {
      float m2 = lm[r][tx], s2 = ls[r][tx];
      float nm = fmaxf(M, m2);
      Ssum = Ssum * __expf(M - nm) + s2 * __expf(m2 - nm);
      M = nm;
    }
    lm[0][tx] = M;
    ls[0][tx] = 1.0f / Ssum;
  }
  __syncthreads();
  const float M    = lm[0][tx];
  const float invS = ls[0][tx];

  for (int i = ty; i < NDIM; i += 8) {
    float x = base[(size_t)i * NDIM + j0 + tx];
    out[(size_t)i * NDIM + j0 + tx] = (__bf16)(__expf(x - M) * invS);
  }
}

// ---------------------------------------------------------------------------
// Persistent chain kernel: 16 workgroups x 8 waves. Each wave owns a 16-row
// strip and half of K; per step it runs 16 accumulating
// v_wmma_f32_16x16x32_bf16 ops where B column 0 carries q[k0..k0+31] (staged
// in LDS as bf16 once per step; non-carrier lanes read from a zeroed LDS pad
// so the B load is branchless). Partial sums for the two K halves combine
// through LDS, then a device-wide phase barrier releases the next step.
// ---------------------------------------------------------------------------
__global__ void __launch_bounds__(BLK) dfa_chain(const __bf16* __restrict__ DB,
                                                 const int* __restrict__ seq,
                                                 const float* __restrict__ f_logit,
                                                 float* qbuf,
                                                 unsigned* bar,
                                                 float* __restrict__ out) {
  const int tid   = threadIdx.x;
  const int lane  = tid & 31;
  const int wave  = tid >> 5;               // 0..7
  const int strip = blockIdx.x * 4 + (wave >> 1);  // 0..63 -> rows [strip*16, +16)
  const int r0    = strip * 16;
  const int kseg  = wave & 1;               // which K half (512 wide)
  const int kbase = kseg * 512;

  // A-fragment addressing per ISA 16-bit A layout:
  //   lanes 0-15 : M=lane,    K chunks {k0..k0+7} and {k0+16..k0+23}
  //   lanes 16-31: M=lane-16, K chunks {k0+8..k0+15} and {k0+24..k0+31}
  const int mrow = r0 + (lane & 15);
  const int koff = (lane & 16) ? 8 : 0;
  // B-fragment: lanes 0-15 hold K {0..7,16..23}, lanes 16-31 hold K {8..15,24..31} (col=lane&15)
  const int bqoff = (lane & 16) ? 8 : 0;
  const bool bcarrier = (lane & 15) == 0;

  float* qb0 = qbuf;
  float* qb1 = qbuf + NDIM;
  unsigned* cnt   = bar;
  unsigned* phase = bar + 1;

  __shared__ __attribute__((aligned(16))) __bf16 qbf[NDIM]; // staged q (2 KB)
  __shared__ __attribute__((aligned(16))) __bf16 zpad[512]; // zeros (1 KB)
  __shared__ float red[8][16];  // cross-wave K-half partials

  // Zero the pad once; every per-step __syncthreads below orders first use.
  for (int i = tid; i < 512; i += BLK) zpad[i] = (__bf16)0.0f;

  // Per-lane branchless B source: carriers walk qbf, others walk zpad.
  const __bf16* bsrc = bcarrier ? (qbf + kbase + bqoff) : zpad;

  for (int t = 0; t < TSEQ; ++t) {
    const int sym = seq[t];
    const __bf16* D    = DB + (size_t)sym * NDIM * NDIM;
    const float* qcur  = (t & 1) ? qb1 : qb0;
    float*       qnext = (t & 1) ? qb0 : qb1;

    // Warm next step's matrix rows for this strip (global_prefetch_b8).
    if (t + 1 < TSEQ) {
      const __bf16* Dn = DB + (size_t)seq[t + 1] * NDIM * NDIM;
      __builtin_prefetch(Dn + (size_t)mrow * NDIM + kbase, 0, 3);
    }

    // Stage q as bf16 into LDS once per step (4 elements per thread).
    {
      const float4 q4 = *(const float4*)(qcur + tid * 4);
      qbf[tid * 4 + 0] = (__bf16)q4.x;
      qbf[tid * 4 + 1] = (__bf16)q4.y;
      qbf[tid * 4 + 2] = (__bf16)q4.z;
      qbf[tid * 4 + 3] = (__bf16)q4.w;
    }
    __syncthreads();

    v8f c = {};
    const __bf16* Drow = D + (size_t)mrow * NDIM;

    #pragma unroll 4
    for (int kk = 0; kk < 512; kk += 32) {
      const int k0 = kbase + kk;

      // A: 16x32 bf16 tile of D (two 16B global loads per lane)
      v8bf lo = *(const v8bf*)(Drow + k0 + koff);
      v8bf hi = *(const v8bf*)(Drow + k0 + koff + 16);
      v16bf a = __builtin_shufflevector(lo, hi,
                  0, 1, 2, 3, 4, 5, 6, 7, 8, 9, 10, 11, 12, 13, 14, 15);

      // B: branchless — all lanes issue two ds_load_b128; non-carriers hit
      // the zero pad (same address across lanes -> LDS broadcast).
      const v8bf* qp = (const v8bf*)(bsrc + kk);
      v8bf blo = qp[0];
      v8bf bhi = qp[2];
      v16bf b = __builtin_shufflevector(blo, bhi,
                  0, 1, 2, 3, 4, 5, 6, 7, 8, 9, 10, 11, 12, 13, 14, 15);

      c = __builtin_amdgcn_wmma_f32_16x16x32_bf16(
              /*neg_a=*/false, a, /*neg_b=*/false, b,
              /*c_mod=*/(short)0, c, /*reuse_a=*/false, /*reuse_b=*/false);
    }

    // Column 0 of C: lane 0 holds rows r0..r0+7, lane 16 holds rows r0+8..r0+15.
    if (lane == 0) {
      #pragma unroll
      for (int m = 0; m < 8; ++m) red[wave][m] = c[m];
    } else if (lane == 16) {
      #pragma unroll
      for (int m = 0; m < 8; ++m) red[wave][8 + m] = c[m];
    }
    __syncthreads();

    // Combine the two K-halves (wave pairs) and publish q_{t+1}.
    if (kseg == 0) {
      if (lane == 0) {
        #pragma unroll
        for (int m = 0; m < 8; ++m)
          qnext[r0 + m] = red[wave][m] + red[wave + 1][m];
      } else if (lane == 16) {
        #pragma unroll
        for (int m = 0; m < 8; ++m)
          qnext[r0 + 8 + m] = red[wave][8 + m] + red[wave + 1][8 + m];
      }
    }

    // Device-wide phase barrier across NWG workgroups.
    __syncthreads();
    if (tid == 0) {
      __builtin_amdgcn_fence(__ATOMIC_RELEASE, "agent");
      unsigned my = __hip_atomic_fetch_add(cnt, 1u, __ATOMIC_ACQ_REL,
                                           __HIP_MEMORY_SCOPE_AGENT);
      if (my == (unsigned)(NWG - 1)) {
        __hip_atomic_store(cnt, 0u, __ATOMIC_RELAXED, __HIP_MEMORY_SCOPE_AGENT);
        __hip_atomic_fetch_add(phase, 1u, __ATOMIC_RELEASE,
                               __HIP_MEMORY_SCOPE_AGENT);
      } else {
        while (__hip_atomic_load(phase, __ATOMIC_ACQUIRE,
                                 __HIP_MEMORY_SCOPE_AGENT) < (unsigned)(t + 1)) {
          __builtin_amdgcn_s_sleep(1);
        }
      }
      __builtin_amdgcn_fence(__ATOMIC_ACQUIRE, "agent");
    }
    __syncthreads();
  }

  // out = sigmoid(f_logit) . q_T   (T even -> final vector is in qb0)
  if (blockIdx.x == 0) {
    __shared__ float racc[BLK];
    float acc = 0.0f;
    for (int i = tid; i < NDIM; i += BLK) {
      float f = 1.0f / (1.0f + __expf(-f_logit[i]));
      acc += f * qb0[i];
    }
    racc[tid] = acc;
    __syncthreads();
    #pragma unroll
    for (int off = BLK / 2; off > 0; off >>= 1) {
      if (tid < off) racc[tid] += racc[tid + off];
      __syncthreads();
    }
    if (tid == 0) out[0] = racc[0];
  }
}

// ---------------------------------------------------------------------------
extern "C" void kernel_launch(void* const* d_in, const int* in_sizes, int n_in,
                              void* d_out, int out_size, void* d_ws, size_t ws_size,
                              hipStream_t stream) {
  const float* delta   = (const float*)d_in[0];  // [S,N,N] f32
  const float* f_logit = (const float*)d_in[1];  // [N] f32
  const int*   seq     = (const int*)d_in[2];    // [T] i32
  float*       out     = (float*)d_out;          // scalar f32

  char* ws = (char*)d_ws;
  __bf16*   DB   = (__bf16*)ws;                         // 64 MB bf16 matrices
  float*    qbuf = (float*)(ws + DB_BYTES);             // 8 KB ping-pong q
  unsigned* bar  = (unsigned*)(ws + DB_BYTES + 8192);   // barrier state

  dfa_init<<<1, BLK, 0, stream>>>(qbuf, bar);
  dfa_softmax<<<dim3(32, SSYM), BLK, 0, stream>>>(delta, DB);
  dfa_chain<<<NWG, BLK, 0, stream>>>(DB, seq, f_logit, qbuf, bar, out);
}
